// S4sequence_37890201485551
// MI455X (gfx1250) — compile-verified, matrix operand order checked by reading, not run on hardware
//
#include <hip/hip_runtime.h>
#include <hip/hip_bf16.h>
#include <math.h>

#define Bz 16
#define Hh 512
#define Ll 2048
#define Nn 64

typedef __attribute__((ext_vector_type(16))) __bf16 v16bf;
typedef __attribute__((ext_vector_type(8)))  float  v8f;

// ---------------- complex helpers ----------------
__device__ __forceinline__ float2 cmul(float2 a, float2 b) {
    return make_float2(a.x * b.x - a.y * b.y, a.x * b.y + a.y * b.x);
}
__device__ __forceinline__ float2 cadd(float2 a, float2 b) { return make_float2(a.x + b.x, a.y + b.y); }
__device__ __forceinline__ float2 csub(float2 a, float2 b) { return make_float2(a.x - b.x, a.y - b.y); }
__device__ __forceinline__ float2 cdiv(float2 a, float2 b) {
    float inv = 1.0f / (b.x * b.x + b.y * b.y);
    return make_float2((a.x * b.x + a.y * b.y) * inv, (a.y * b.x - a.x * b.y) * inv);
}

// ---------------- 1) per-batch mean/var partials ----------------
__global__ __launch_bounds__(256) void s4_reduce_partial(const float* __restrict__ x,
                                                         float* __restrict__ part) {
    int b = blockIdx.y;
    int blk = blockIdx.x;               // 64 blocks per batch
    int t = threadIdx.x;
    const float* xb = x + (size_t)b * Ll * Hh;
    float s = 0.f, s2 = 0.f;
    for (int i = blk * 256 + t; i < Ll * Hh; i += 64 * 256) {
        float v = xb[i];
        s += v; s2 += v * v;
    }
    __shared__ float sh0[256], sh1[256];
    sh0[t] = s; sh1[t] = s2;
    __syncthreads();
    for (int off = 128; off > 0; off >>= 1) {
        if (t < off) { sh0[t] += sh0[t + off]; sh1[t] += sh1[t + off]; }
        __syncthreads();
    }
    if (t == 0) {
        part[(b * 64 + blk) * 2 + 0] = sh0[0];
        part[(b * 64 + blk) * 2 + 1] = sh1[0];
    }
}

__global__ __launch_bounds__(64) void s4_reduce_final(const float* __restrict__ part,
                                                      float* __restrict__ stats) {
    int b = blockIdx.x;
    int t = threadIdx.x;                // 64 threads
    __shared__ float sh0[64], sh1[64];
    sh0[t] = part[(b * 64 + t) * 2 + 0];
    sh1[t] = part[(b * 64 + t) * 2 + 1];
    __syncthreads();
    for (int off = 32; off > 0; off >>= 1) {
        if (t < off) { sh0[t] += sh0[t + off]; sh1[t] += sh1[t + off]; }
        __syncthreads();
    }
    if (t == 0) {
        const float invn = 1.0f / (float)(Ll * Hh);
        float mu = sh0[0] * invn;
        float var = sh1[0] * invn - mu * mu;
        stats[b * 2 + 0] = mu;
        stats[b * 2 + 1] = rsqrtf(var + 1e-5f);
    }
}

// ---------------- 2) S4 DPLR frequency-domain kernel ----------------
__global__ __launch_bounds__(256) void s4_freq_kernel(
    const float* __restrict__ lre, const float* __restrict__ lim,
    const float* __restrict__ pre, const float* __restrict__ pim,
    const float* __restrict__ bre, const float* __restrict__ bim,
    const float* __restrict__ cre, const float* __restrict__ cim,
    float2* __restrict__ kern) {
    int idx = blockIdx.x * 256 + threadIdx.x;     // H*L threads
    int l = idx & (Ll - 1);
    int h = idx >> 11;                            // L = 2048
    float ang = -6.283185307179586f * (float)l / (float)Ll;
    float oi, orr;
    __sincosf(ang, &oi, &orr);                    // Omega = orr + i*oi
    float pr = 1.0f + orr, pi2 = oi;              // 1 + Omega
    float inv = 1.0f / (pr * pr + pi2 * pi2);
    float2 c = make_float2(2.0f * pr * inv, -2.0f * pi2 * inv);   // 2/(1+Omega)
    float mr = 1.0f - orr, mi = -oi;              // 1 - Omega
    const float scale = 2.0f / 0.001f;            // 2 / STEP
    float gr = (mr * pr + mi * pi2) * inv * scale;
    float gi = (mi * pr - mr * pi2) * inv * scale;
    float2 k0 = {0.f, 0.f}, k1 = {0.f, 0.f}, k2 = {0.f, 0.f}, k3 = {0.f, 0.f};
    const int base = h * Nn;
    #pragma unroll 4
    for (int n = 0; n < Nn; n++) {
        float lr = lre[base + n], li = lim[base + n];
        float dr = gr - lr, di = gi - li;
        float rinv = 1.0f / (dr * dr + di * di);
        float2 r = make_float2(dr * rinv, -di * rinv);            // 1/(g-Lam)
        float Pr = pre[base + n], Pi = pim[base + n];
        float Br = bre[base + n], Bi = bim[base + n];
        float Cr = cre[base + n], Ci = cim[base + n];
        float2 cC = make_float2(Cr, -Ci);
        float2 cP = make_float2(Pr, -Pi);
        float2 Bc = make_float2(Br, Bi);
        float2 Pc = make_float2(Pr, Pi);
        k0 = cadd(k0, cmul(cmul(cC, Bc), r));
        k1 = cadd(k1, cmul(cmul(cC, Pc), r));
        k2 = cadd(k2, cmul(cmul(cP, Bc), r));
        k3 = cadd(k3, cmul(cmul(cP, Pc), r));
    }
    float2 den = make_float2(1.0f + k3.x, k3.y);
    float2 tt = cdiv(cmul(k1, k2), den);
    float2 res = cmul(c, csub(k0, tt));
    kern[(size_t)h * Ll + l] = res;
}

// ---------------- 3) LayerNorm + transpose [B,L,H] -> [B,H,L] ----------------
__global__ __launch_bounds__(256) void s4_ln_transpose(
    const float* __restrict__ x, const float* __restrict__ gamma,
    const float* __restrict__ beta, const float* __restrict__ stats,
    float* __restrict__ xT) {
    __shared__ float tile[32][33];
    int b = blockIdx.z;
    int l0 = blockIdx.y * 32;
    int h0 = blockIdx.x * 32;
    float mu = stats[b * 2 + 0];
    float rstd = stats[b * 2 + 1];
    int tx = threadIdx.x, ty = threadIdx.y;       // (32,8)
    #pragma unroll
    for (int i = 0; i < 4; i++) {
        int l = l0 + ty + 8 * i;
        int h = h0 + tx;
        size_t gi = (size_t)l * Hh + h;
        float v = x[(size_t)b * Ll * Hh + gi];
        tile[ty + 8 * i][tx] = (v - mu) * rstd * gamma[gi] + beta[gi];
    }
    __syncthreads();
    #pragma unroll
    for (int i = 0; i < 4; i++) {
        int h = h0 + ty + 8 * i;
        int l = l0 + tx;
        xT[((size_t)b * Hh + h) * Ll + l] = tile[tx][ty + 8 * i];
    }
}

// ---------------- 4) FFT-conv (2048-pt Stockham in LDS, in place per row) ----
__device__ __forceinline__ void fft2048_stages(float2 (*buf)[2048], int& src, int t) {
    #pragma unroll 1
    for (int s = 0; s < 11; s++) {
        int m = 1 << s;
        int l2 = 1024 >> s;
        int dst = src ^ 1;
        #pragma unroll
        for (int jj = 0; jj < 4; jj++) {
            int j = t + jj * 256;
            int p = j >> s;
            int q = j & (m - 1);
            float2 a = buf[src][j];
            float2 bb = buf[src][j + 1024];
            float ang = -3.14159265358979f * (float)p / (float)l2;
            float ws, wc;
            __sincosf(ang, &ws, &wc);
            float2 sum = make_float2(a.x + bb.x, a.y + bb.y);
            float2 dif = make_float2(a.x - bb.x, a.y - bb.y);
            float2 dw = make_float2(dif.x * wc - dif.y * ws, dif.x * ws + dif.y * wc);
            int i0 = ((p * m) << 1) + q;
            buf[dst][i0] = sum;
            buf[dst][i0 + m] = dw;
        }
        __syncthreads();
        src = dst;
    }
}

__global__ __launch_bounds__(256) void s4_fft_conv(float* __restrict__ xT,
                                                   const float2* __restrict__ kern) {
    __shared__ float2 buf[2][2048];               // 32 KB
    int bh = blockIdx.x;                          // b*H + h
    int h = bh & (Hh - 1);
    float* row = xT + (size_t)bh * Ll;
    int t = threadIdx.x;
    for (int i = t; i < Ll; i += 256) buf[0][i] = make_float2(row[i], 0.f);
    __syncthreads();
    int src = 0;
    fft2048_stages(buf, src, t);                  // forward FFT
    const float2* kr = kern + (size_t)h * Ll;
    for (int i = t; i < Ll; i += 256) {
        float2 u = buf[src][i];
        float2 kk = kr[i];
        float2 y = cmul(kk, u);
        buf[src][i] = make_float2(y.x, -y.y);     // conj -> inverse via forward FFT
    }
    __syncthreads();
    fft2048_stages(buf, src, t);                  // "inverse" FFT
    const float s = 1.0f / 2048.0f;
    for (int i = t; i < Ll; i += 256) row[i] = buf[src][i].x * s;
}

// ---------------- 5) GELU + transpose back [B,H,L] -> [B,L,H] bf16 ----------
__global__ __launch_bounds__(256) void s4_gelu_transpose(const float* __restrict__ yT,
                                                         __bf16* __restrict__ yg) {
    __shared__ float tile[32][33];
    int b = blockIdx.z;
    int h0 = blockIdx.y * 32;
    int l0 = blockIdx.x * 32;
    int tx = threadIdx.x, ty = threadIdx.y;       // (32,8)
    #pragma unroll
    for (int i = 0; i < 4; i++) {
        int h = h0 + ty + 8 * i;
        tile[ty + 8 * i][tx] = yT[((size_t)b * Hh + h) * Ll + l0 + tx];
    }
    __syncthreads();
    #pragma unroll
    for (int i = 0; i < 4; i++) {
        int l = l0 + ty + 8 * i;
        int h = h0 + tx;
        float v = tile[tx][ty + 8 * i];
        float g = 0.5f * v * (1.0f + erff(v * 0.70710678f));   // exact GELU
        yg[((size_t)b * Ll + l) * Hh + h] = (__bf16)g;
    }
}

// ---------------- 6) dual-GEMM GLU projection with bf16 WMMA ----------------
// Fragment gather per CDNA5 16-bit A layout: lane = (row%16) + 16*hiK,
// VGPR v holds K-pair k = (v<4?0:16) + hi*8 + (v%4)*2 + {0,1}. Row stride 36.
__device__ __forceinline__ v16bf load_frag36(const __bf16* base, int lane) {
    int rr = lane & 15;
    int hi = (lane >> 4) & 1;
    const __bf16* rowp = base + rr * 36;
    v16bf f;
    #pragma unroll
    for (int v = 0; v < 8; v++) {
        int k = ((v >> 2) << 4) + (hi << 3) + ((v & 3) << 1);   // even K of the pair
        union { unsigned int u; __bf16 h[2]; } cv;
        cv.u = *(const unsigned int*)(rowp + k);                // 4B-aligned LDS load
        f[2 * v]     = cv.h[0];
        f[2 * v + 1] = cv.h[1];
    }
    return f;
}

__global__ __launch_bounds__(256) void s4_glu_gemm(
    const __bf16* __restrict__ yg, const float* __restrict__ W1,
    const float* __restrict__ b1, const float* __restrict__ W2,
    const float* __restrict__ b2, const float* __restrict__ xskip,
    float* __restrict__ out) {
    __shared__ __bf16 As[128 * 36];
    __shared__ __bf16 Bs1[64 * 36];
    __shared__ __bf16 Bs2[64 * 36];
    const int m0 = blockIdx.x * 128;
    const int n0 = blockIdx.y * 64;
    const int t = threadIdx.x;
    const int lane = t & 31;
    const int wave = t >> 5;
    const int wm = wave & 3;                       // 4 wave-rows of 32
    const int wn = wave >> 2;                      // 2 wave-cols of 32
    const v8f vzero = {0.f, 0.f, 0.f, 0.f, 0.f, 0.f, 0.f, 0.f};
    v8f acc1[2][2], acc2[2][2];
    #pragma unroll
    for (int i = 0; i < 2; i++)
        #pragma unroll
        for (int j = 0; j < 2; j++) { acc1[i][j] = vzero; acc2[i][j] = vzero; }

    for (int k0 = 0; k0 < Hh; k0 += 32) {
        // stage A (bf16 activations): 128x32
        #pragma unroll
        for (int i = 0; i < 4; i++) {
            int id = t + 256 * i;
            int r = id >> 3;
            int c4 = (id & 7) << 2;
            uint2 v = *(const uint2*)(yg + (size_t)(m0 + r) * Hh + k0 + c4);
            *(uint2*)(&As[r * 36 + c4]) = v;
        }
        // stage both weight tiles (f32 -> bf16): 64x32 each, Bs[n][k] = W[n][k]
        #pragma unroll
        for (int i = 0; i < 2; i++) {
            int id = t + 256 * i;
            int r = id >> 3;
            int c4 = (id & 7) << 2;
            float4 w1v = *(const float4*)(W1 + (size_t)(n0 + r) * Hh + k0 + c4);
            float4 w2v = *(const float4*)(W2 + (size_t)(n0 + r) * Hh + k0 + c4);
            __bf16* p1 = &Bs1[r * 36 + c4];
            __bf16* p2 = &Bs2[r * 36 + c4];
            p1[0] = (__bf16)w1v.x; p1[1] = (__bf16)w1v.y; p1[2] = (__bf16)w1v.z; p1[3] = (__bf16)w1v.w;
            p2[0] = (__bf16)w2v.x; p2[1] = (__bf16)w2v.y; p2[2] = (__bf16)w2v.z; p2[3] = (__bf16)w2v.w;
        }
        __syncthreads();

        v16bf a0  = load_frag36(&As[(wm * 32 + 0)  * 36], lane);
        v16bf a1  = load_frag36(&As[(wm * 32 + 16) * 36], lane);
        v16bf b10 = load_frag36(&Bs1[(wn * 32 + 0)  * 36], lane);
        v16bf b11 = load_frag36(&Bs1[(wn * 32 + 16) * 36], lane);
        v16bf b20 = load_frag36(&Bs2[(wn * 32 + 0)  * 36], lane);
        v16bf b21 = load_frag36(&Bs2[(wn * 32 + 16) * 36], lane);

        acc1[0][0] = __builtin_amdgcn_wmma_f32_16x16x32_bf16(false, a0, false, b10, (short)0, acc1[0][0], false, false);
        acc1[0][1] = __builtin_amdgcn_wmma_f32_16x16x32_bf16(false, a0, false, b11, (short)0, acc1[0][1], false, false);
        acc1[1][0] = __builtin_amdgcn_wmma_f32_16x16x32_bf16(false, a1, false, b10, (short)0, acc1[1][0], false, false);
        acc1[1][1] = __builtin_amdgcn_wmma_f32_16x16x32_bf16(false, a1, false, b11, (short)0, acc1[1][1], false, false);
        acc2[0][0] = __builtin_amdgcn_wmma_f32_16x16x32_bf16(false, a0, false, b20, (short)0, acc2[0][0], false, false);
        acc2[0][1] = __builtin_amdgcn_wmma_f32_16x16x32_bf16(false, a0, false, b21, (short)0, acc2[0][1], false, false);
        acc2[1][0] = __builtin_amdgcn_wmma_f32_16x16x32_bf16(false, a1, false, b20, (short)0, acc2[1][0], false, false);
        acc2[1][1] = __builtin_amdgcn_wmma_f32_16x16x32_bf16(false, a1, false, b21, (short)0, acc2[1][1], false, false);
        __syncthreads();
    }

    // epilogue: h*sigmoid(g) + skip, fused bias
    const int nn = lane & 15;
    const int hi = lane >> 4;
    #pragma unroll
    for (int mi = 0; mi < 2; mi++) {
        #pragma unroll
        for (int ni = 0; ni < 2; ni++) {
            int n = n0 + wn * 32 + ni * 16 + nn;
            float bb1 = b1[n], bb2 = b2[n];
            int mbase = m0 + wm * 32 + mi * 16 + (hi << 3);
            v8f h1 = acc1[mi][ni];
            v8f h2 = acc2[mi][ni];
            #pragma unroll
            for (int r = 0; r < 8; r++) {
                int m = mbase + r;
                float hv = h1[r] + bb1;
                float gv = h2[r] + bb2;
                float sg = 1.0f / (1.0f + __expf(-gv));
                size_t idx = (size_t)m * Hh + n;
                out[idx] = hv * sg + xskip[idx];
            }
        }
    }
}

// ---------------- host launcher ----------------
extern "C" void kernel_launch(void* const* d_in, const int* in_sizes, int n_in,
                              void* d_out, int out_size, void* d_ws, size_t ws_size,
                              hipStream_t stream) {
    (void)in_sizes; (void)n_in; (void)out_size; (void)ws_size;
    const float* x      = (const float*)d_in[0];
    const float* gamma  = (const float*)d_in[1];
    const float* beta   = (const float*)d_in[2];
    const float* Lam_re = (const float*)d_in[3];
    const float* Lam_im = (const float*)d_in[4];
    const float* P_re   = (const float*)d_in[5];
    const float* P_im   = (const float*)d_in[6];
    const float* B_re   = (const float*)d_in[7];
    const float* B_im   = (const float*)d_in[8];
    const float* C_re   = (const float*)d_in[9];
    const float* C_im   = (const float*)d_in[10];
    const float* W1     = (const float*)d_in[11];
    const float* b1     = (const float*)d_in[12];
    const float* W2     = (const float*)d_in[13];
    const float* b2     = (const float*)d_in[14];
    float* out = (float*)d_out;

    // workspace layout (bytes)
    char* ws = (char*)d_ws;
    float*  part  = (float*)(ws);                                  //   8 KB
    float*  stats = (float*)(ws + 8192);                           //  128 B
    float2* kern  = (float2*)(ws + 16384);                         //   8 MB
    float*  xT    = (float*)(ws + 16384 + 8u * 1024 * 1024);       //  64 MB (xn, then y, in place)
    __bf16* yg    = (__bf16*)(ws + 16384 + 8u * 1024 * 1024 + 64u * 1024 * 1024); // 32 MB

    // 1) LayerNorm statistics (two-stage reduction)
    s4_reduce_partial<<<dim3(64, Bz), 256, 0, stream>>>(x, part);
    s4_reduce_final<<<Bz, 64, 0, stream>>>(part, stats);

    // 2) S4 DPLR frequency kernel  [H, L]
    s4_freq_kernel<<<(Hh * Ll) / 256, 256, 0, stream>>>(
        Lam_re, Lam_im, P_re, P_im, B_re, B_im, C_re, C_im, kern);

    // 3) LayerNorm + transpose to [B, H, L]
    s4_ln_transpose<<<dim3(Hh / 32, Ll / 32, Bz), dim3(32, 8), 0, stream>>>(
        x, gamma, beta, stats, xT);

    // 4) FFT * kern * IFFT per (b,h) row, in place
    s4_fft_conv<<<Bz * Hh, 256, 0, stream>>>(xT, kern);

    // 5) GELU + transpose back to [B, L, H] as bf16
    s4_gelu_transpose<<<dim3(Ll / 32, Hh / 32, Bz), dim3(32, 8), 0, stream>>>(xT, yg);

    // 6) fused dual-GEMM GLU + residual (bf16 WMMA, f32 accumulate)
    s4_glu_gemm<<<dim3((Bz * Ll) / 128, Hh / 64), 256, 0, stream>>>(
        yg, W1, b1, W2, b2, x, out);
}
